// STGCNBlock_87162066305605
// MI455X (gfx1250) — matrix operand order repaired and checked by reading, not compiled, not run on hardware
//
#include <hip/hip_runtime.h>

#define EPS 1e-5f

typedef __attribute__((ext_vector_type(16))) __bf16 v16bf;
typedef __attribute__((ext_vector_type(8)))  __bf16 v8bf;
typedef __attribute__((ext_vector_type(4)))  __bf16 v4bf;
typedef __attribute__((ext_vector_type(8)))  float  v8f;

__device__ __forceinline__ __bf16 f2bf(float f) { return (__bf16)f; }
__device__ __forceinline__ float bf2f(__bf16 h) { return (float)h; }

__device__ __forceinline__ v8f vzero8() {
  v8f z;
#pragma unroll
  for (int i = 0; i < 8; i++) z[i] = 0.0f;
  return z;
}

__device__ __forceinline__ v8f wmma_bf16(v16bf a, v16bf b, v8f c) {
  return __builtin_amdgcn_wmma_f32_16x16x32_bf16(false, a, false, b, (short)0, c,
                                                 false, false);
}

// A fragment (16x32 bf16): lane holds row m; k = kb*32 + {half*8..+7, +16..+23}
__device__ __forceinline__ v16bf frag_a(const __bf16* row, int kb, int half) {
  const __bf16* p = row + kb * 32 + half * 8;
  v8bf lo = *(const v8bf*)p;
  v8bf hi = *(const v8bf*)(p + 16);
  return __builtin_shufflevector(lo, hi, 0, 1, 2, 3, 4, 5, 6, 7, 8, 9, 10, 11,
                                 12, 13, 14, 15);
}
// B fragment (32x16 bf16): lane holds col n; k = kb*32 + {half*8..+7, +16..+23}
// -> two 16-byte loads when B is stored as [n][k].
__device__ __forceinline__ v16bf frag_b(const __bf16* row, int kb, int half) {
  const __bf16* p = row + kb * 32 + half * 8;
  v8bf lo = *(const v8bf*)p;
  v8bf hi = *(const v8bf*)(p + 16);
  return __builtin_shufflevector(lo, hi, 0, 1, 2, 3, 4, 5, 6, 7, 8, 9, 10, 11,
                                 12, 13, 14, 15);
}

// ---------------------------------------------------------------------------
// K1: GCN — per (b,c). Weights in A (loop-invariant frags live in VGPRs),
// activations in B => D arrives transposed; lanes write 8 contiguous bf16.
//   stage1: M1^T[w'][t] = adj_t(A) x X^T(B)   -> strip2[t][w']
//   stage2: H^T [w][t]  = gw_t (A) x M1^T(B)  -> strip [t][w] -> hraw (bf16)
// BN1 stats accumulated in the coalesced store loop.
// ---------------------------------------------------------------------------
__global__ void __launch_bounds__(256)
gcn_kernel(const float* __restrict__ x, const float* __restrict__ adj,
           const float* __restrict__ gw, __bf16* __restrict__ hraw,
           float* __restrict__ partial) {
  __shared__ __align__(16) __bf16 adj_t[64][64];     // [m=w'][k=v]
  __shared__ __align__(16) __bf16 gw_t[64][64];      // [m=w][k=w']
  __shared__ __align__(16) __bf16 strip[8][16][64];  // per-wave [t][k]
  __shared__ __align__(16) __bf16 strip2[8][16][64]; // per-wave [t][w']
  __shared__ float redS[8][4][8][8];
  __shared__ float redQ[8][4][8][8];
  const int b = blockIdx.x >> 6, c = blockIdx.x & 63;
  const int tid = threadIdx.x, wave = tid >> 5, lane = tid & 31;
  const int mrow = lane & 15, half = lane >> 4;

  for (int idx = tid; idx < 4096; idx += 256) {
    int w = idx >> 6, v = idx & 63;
    adj_t[w][v] = f2bf(adj[(v * 64 + w) * 64 + c]);
    gw_t[w][v] = f2bf(gw[(v * 64 + w) * 64 + c]);
  }
  __syncthreads();

  const float* xrow = x + (size_t)(b * 64 + c) * 65536;
  __bf16* hrow = hraw + (size_t)(b * 64 + c) * 65536;

  float sAcc[8], qAcc[8];
#pragma unroll
  for (int j = 0; j < 8; j++) { sAcc[j] = 0.f; qAcc[j] = 0.f; }

  for (int t0 = wave * 16; t0 < 1024; t0 += 128) {
    // stage X strip [16][64] into LDS as bf16
    for (int j = lane; j < 256; j += 32) {
      int r = j >> 4, c4 = (j & 15) * 4;
      float4 val = *(const float4*)(xrow + (t0 + r) * 64 + c4);
      v4bf pk;
      pk[0] = f2bf(val.x); pk[1] = f2bf(val.y);
      pk[2] = f2bf(val.z); pk[3] = f2bf(val.w);
      *(v4bf*)(&strip[wave][r][c4]) = pk;
    }
    // ---- stage 1
    v8f acc0[4];
#pragma unroll
    for (int mt = 0; mt < 4; mt++) acc0[mt] = vzero8();
#pragma unroll
    for (int kb = 0; kb < 2; kb++) {
      v16bf bfrag = frag_b(&strip[wave][mrow][0], kb, half);
#pragma unroll
      for (int mt = 0; mt < 4; mt++) {
        v16bf afrag = frag_a(&adj_t[mt * 16 + mrow][0], kb, half);
        acc0[mt] = wmma_bf16(afrag, bfrag, acc0[mt]);
      }
    }
#pragma unroll
    for (int mt = 0; mt < 4; mt++) {
      v8bf pk;
#pragma unroll
      for (int r = 0; r < 8; r++) pk[r] = f2bf(acc0[mt][r]);
      *(v8bf*)(&strip2[wave][mrow][mt * 16 + half * 8]) = pk;
    }
    // ---- stage 2
    v8f acc1[4];
#pragma unroll
    for (int mt = 0; mt < 4; mt++) acc1[mt] = vzero8();
#pragma unroll
    for (int kb = 0; kb < 2; kb++) {
      v16bf bfrag = frag_b(&strip2[wave][mrow][0], kb, half);
#pragma unroll
      for (int mt = 0; mt < 4; mt++) {
        v16bf afrag = frag_a(&gw_t[mt * 16 + mrow][0], kb, half);
        acc1[mt] = wmma_bf16(afrag, bfrag, acc1[mt]);
      }
    }
#pragma unroll
    for (int mt = 0; mt < 4; mt++) {
      v8bf pk;
#pragma unroll
      for (int r = 0; r < 8; r++) pk[r] = f2bf(acc1[mt][r]);
      *(v8bf*)(&strip[wave][mrow][mt * 16 + half * 8]) = pk;
    }
    // coalesced global store + BN1 stats (fixed 8-channel group per lane)
    for (int j = lane; j < 128; j += 32) {
      int r = j >> 3, c8 = (j & 7) * 8;
      v8bf hv = *(const v8bf*)&strip[wave][r][c8];
      *(v8bf*)(hrow + (t0 + r) * 64 + c8) = hv;
#pragma unroll
      for (int jj = 0; jj < 8; jj++) {
        float v = bf2f(hv[jj]);
        sAcc[jj] += v;
        qAcc[jj] += v * v;
      }
    }
  }
#pragma unroll
  for (int jj = 0; jj < 8; jj++) {
    redS[wave][lane >> 3][lane & 7][jj] = sAcc[jj];
    redQ[wave][lane >> 3][lane & 7][jj] = qAcc[jj];
  }
  __syncthreads();
  if (tid < 64) { // channel = tid = (tid>>3)*8 + (tid&7)
    int g = tid >> 3, jj = tid & 7;
    float s = 0.f, q = 0.f;
    for (int wv = 0; wv < 8; wv++)
      for (int u = 0; u < 4; u++) {
        s += redS[wv][u][g][jj];
        q += redQ[wv][u][g][jj];
      }
    partial[blockIdx.x * 128 + tid] = s;
    partial[blockIdx.x * 128 + 64 + tid] = q;
  }
}

// ---------------------------------------------------------------------------
// BN2 stats partial: y = relu(sc1*h + sh1). Vectorized: one v8bf load per 8
// elements, lane owns a fixed 8-channel group; deterministic LDS reduce.
// ---------------------------------------------------------------------------
__global__ void __launch_bounds__(256)
stats_partial(const __bf16* __restrict__ data, float* __restrict__ partial,
              const float* __restrict__ sc, const float* __restrict__ sh) {
  __shared__ float redS[32][8][8];
  __shared__ float redQ[32][8][8];
  const int tid = threadIdx.x;
  const int g = tid & 7, rg = tid >> 3; // 32 row-groups
  const int c8 = g * 8;
  float a[8], o[8], s[8], q[8];
#pragma unroll
  for (int j = 0; j < 8; j++) {
    a[j] = sc[c8 + j];
    o[j] = sh[c8 + j];
    s[j] = 0.f;
    q[j] = 0.f;
  }
  const size_t base = (size_t)blockIdx.x * 1024;
  for (int r = rg; r < 1024; r += 32) {
    v8bf hv = *(const v8bf*)(data + (base + r) * 64 + c8);
#pragma unroll
    for (int j = 0; j < 8; j++) {
      float v = fmaxf(a[j] * bf2f(hv[j]) + o[j], 0.0f);
      s[j] += v;
      q[j] += v * v;
    }
  }
#pragma unroll
  for (int j = 0; j < 8; j++) {
    redS[rg][g][j] = s[j];
    redQ[rg][g][j] = q[j];
  }
  __syncthreads();
  if (tid < 64) {
    int gg = tid >> 3, jj = tid & 7;
    float ss = 0.f, qq = 0.f;
    for (int i = 0; i < 32; i++) {
      ss += redS[i][gg][jj];
      qq += redQ[i][gg][jj];
    }
    partial[blockIdx.x * 128 + tid] = ss;
    partial[blockIdx.x * 128 + 64 + tid] = qq;
  }
}

__global__ void stats_finalize(const float* __restrict__ partial,
                               const float* __restrict__ gamma,
                               const float* __restrict__ beta,
                               float* __restrict__ sc, float* __restrict__ sh) {
  const int w = threadIdx.x; // 64 threads
  float s = 0.0f, q = 0.0f;
  for (int i = 0; i < 512; i++) {
    s += partial[i * 128 + w];
    q += partial[i * 128 + 64 + w];
  }
  const float invN = 1.0f / 524288.0f; // B*T*64
  float mean = s * invN;
  float var = q * invN - mean * mean;
  float scale = gamma[w] * rsqrtf(var + EPS);
  sc[w] = scale;
  sh[w] = beta[w] - mean * scale;
}

// ---------------------------------------------------------------------------
// K-conv: per (b,c): Z^T[o][t] = sum_{dt} cw_t[dt](A) x f(H)(B) ; +cb; bf16 Z.
// f = relu(bn2(relu(bn1(.)))). BN3 stats in the coalesced store loop.
// ---------------------------------------------------------------------------
__global__ void __launch_bounds__(256)
conv_kernel(const __bf16* __restrict__ hraw, const float* __restrict__ cw,
            const float* __restrict__ cb,
            const float* __restrict__ sc1, const float* __restrict__ sh1,
            const float* __restrict__ sc2, const float* __restrict__ sh2,
            __bf16* __restrict__ z, float* __restrict__ partial) {
  __shared__ __align__(16) __bf16 cw_t[9][64][64];   // [dt][m=o][k=i] (72 KB)
  __shared__ __align__(16) __bf16 in_s[136][64];     // halo 4 each side
  __shared__ __align__(16) __bf16 zstrip[8][16][64]; // per-wave [t][o]
  __shared__ float bnp[4][64];
  __shared__ float cbs[64];
  __shared__ float redS[8][4][8][8];
  __shared__ float redQ[8][4][8][8];
  const int b = blockIdx.x >> 6, c = blockIdx.x & 63;
  const int tid = threadIdx.x, wave = tid >> 5, lane = tid & 31;
  const int mrow = lane & 15, half = lane >> 4;

  for (int idx = tid; idx < 9 * 4096; idx += 256) {
    int dt = idx >> 12, rem = idx & 4095;
    int o = rem >> 6, i = rem & 63;
    cw_t[dt][o][i] = f2bf(cw[(o * 64 + i) * 9 + dt]); // conv_w[o][i][dt][0]
  }
  if (tid < 64) {
    bnp[0][tid] = sc1[tid];
    bnp[1][tid] = sh1[tid];
    bnp[2][tid] = sc2[tid];
    bnp[3][tid] = sh2[tid];
    cbs[tid] = cb[tid];
  }

  const __bf16* hbase = hraw + (size_t)(b * 64 + c) * 65536;
  __bf16* zbase = z + (size_t)(b * 64 + c) * 65536;

  float sAcc[8], qAcc[8];
#pragma unroll
  for (int j = 0; j < 8; j++) { sAcc[j] = 0.f; qAcc[j] = 0.f; }

  for (int t0c = 0; t0c < 1024; t0c += 128) {
    __syncthreads(); // cw_t/bnp ready (1st iter); in_s free of prior readers
    for (int qd = tid; qd < 1088; qd += 256) { // 136*64/8 groups of 8
      int r = qd >> 3, c8 = (qd & 7) * 8;
      int t = t0c - 4 + r;
      v8bf pk;
      if (t >= 0 && t < 1024) {
        v8bf hv = *(const v8bf*)(hbase + t * 64 + c8);
#pragma unroll
        for (int j = 0; j < 8; j++) {
          float v = bf2f(hv[j]);
          v = fmaxf(bnp[0][c8 + j] * v + bnp[1][c8 + j], 0.0f);
          v = fmaxf(bnp[2][c8 + j] * v + bnp[3][c8 + j], 0.0f);
          pk[j] = f2bf(v);
        }
      } else {
#pragma unroll
        for (int j = 0; j < 8; j++) pk[j] = f2bf(0.0f);
      }
      *(v8bf*)(&in_s[r][c8]) = pk;
    }
    __syncthreads();

    v8f acc[4];
#pragma unroll
    for (int mt = 0; mt < 4; mt++) acc[mt] = vzero8();
    const int rrow = wave * 16 + mrow; // + dt gives in_s row (halo-shifted)
    for (int dt = 0; dt < 9; dt++) {
#pragma unroll
      for (int kb = 0; kb < 2; kb++) {
        v16bf bfrag = frag_b(&in_s[rrow + dt][0], kb, half);
#pragma unroll
        for (int mt = 0; mt < 4; mt++) {
          v16bf afrag = frag_a(&cw_t[dt][mt * 16 + mrow][0], kb, half);
          acc[mt] = wmma_bf16(afrag, bfrag, acc[mt]);
        }
      }
    }
    // D[m=o][n=t]: 8 contiguous o per lane -> packed cvt + b128 store
#pragma unroll
    for (int mt = 0; mt < 4; mt++) {
      v8bf pk;
#pragma unroll
      for (int r = 0; r < 8; r++)
        pk[r] = f2bf(acc[mt][r] + cbs[mt * 16 + half * 8 + r]);
      *(v8bf*)(&zstrip[wave][mrow][mt * 16 + half * 8]) = pk;
    }
    const int t0 = t0c + wave * 16;
    for (int j = lane; j < 128; j += 32) {
      int r = j >> 3, c8 = (j & 7) * 8;
      v8bf hv = *(const v8bf*)&zstrip[wave][r][c8];
      *(v8bf*)(zbase + (t0 + r) * 64 + c8) = hv;
#pragma unroll
      for (int jj = 0; jj < 8; jj++) {
        float v = bf2f(hv[jj]);
        sAcc[jj] += v;
        qAcc[jj] += v * v;
      }
    }
  }
#pragma unroll
  for (int jj = 0; jj < 8; jj++) {
    redS[wave][lane >> 3][lane & 7][jj] = sAcc[jj];
    redQ[wave][lane >> 3][lane & 7][jj] = qAcc[jj];
  }
  __syncthreads();
  if (tid < 64) {
    int g = tid >> 3, jj = tid & 7;
    float s = 0.f, q = 0.f;
    for (int wv = 0; wv < 8; wv++)
      for (int u = 0; u < 4; u++) {
        s += redS[wv][u][g][jj];
        q += redQ[wv][u][g][jj];
      }
    partial[blockIdx.x * 128 + tid] = s;
    partial[blockIdx.x * 128 + 64 + tid] = q;
  }
}

// ---------------------------------------------------------------------------
// Final: per (b,t): out^T[v][ch] = x_t(A) x res_w(B); lane owns one ch.
// out[ch][v] = relu( bn3(Z[v][ch]) + res + rb ). Vector global loads,
// transposed LDS scatter on store side only; coalesced float4 writeback.
// ---------------------------------------------------------------------------
__global__ void __launch_bounds__(128)
final_kernel(const float* __restrict__ x, const float* __restrict__ rw,
             const float* __restrict__ rb, const __bf16* __restrict__ zsrc,
             const float* __restrict__ sc3, const float* __restrict__ sh3,
             float* __restrict__ out) {
  __shared__ __align__(16) __bf16 rw_s[64][64]; // [n=o][k=cin]
  __shared__ __align__(16) __bf16 x_t[64][64];  // [m=v][k=cin]
  __shared__ __align__(16) __bf16 z_t[64][64];  // [ch][v] (bf16)
  __shared__ __align__(16) float out_s[64][64]; // [ch][v]
  const int bI = blockIdx.x >> 10, t = blockIdx.x & 1023;
  const int tid = threadIdx.x, wave = tid >> 5, lane = tid & 31;
  const int mrow = lane & 15, half = lane >> 4;

  const size_t gbase = ((size_t)(bI * 64) * 1024 + t) * 64;
  // res_w: vector load + vector store (both sides contiguous)
  for (int qd = tid; qd < 1024; qd += 128) {
    int r = qd >> 4, c4 = (qd & 15) * 4;
    float4 wv = *(const float4*)(rw + r * 64 + c4);
    v4bf pk;
    pk[0] = f2bf(wv.x); pk[1] = f2bf(wv.y);
    pk[2] = f2bf(wv.z); pk[3] = f2bf(wv.w);
    *(v4bf*)(&rw_s[r][c4]) = pk;
  }
  // x: float4 load, transposed bf16 scatter
  for (int qd = tid; qd < 1024; qd += 128) {
    int r = qd >> 4, c4 = (qd & 15) * 4;
    float4 wv = *(const float4*)(x + gbase + (size_t)r * 65536 + c4);
    x_t[c4 + 0][r] = f2bf(wv.x);
    x_t[c4 + 1][r] = f2bf(wv.y);
    x_t[c4 + 2][r] = f2bf(wv.z);
    x_t[c4 + 3][r] = f2bf(wv.w);
  }
  // Z: v8bf load, transposed bf16 scatter (no convert)
  for (int qd = tid; qd < 512; qd += 128) {
    int r = qd >> 3, c8 = (qd & 7) * 8;
    v8bf hv = *(const v8bf*)(zsrc + gbase + (size_t)r * 65536 + c8);
#pragma unroll
    for (int jj = 0; jj < 8; jj++) z_t[c8 + jj][r] = hv[jj];
  }
  __syncthreads();

  const int ch = wave * 16 + mrow; // lane's output channel (n)
  const float rbv = rb[ch], s3 = sc3[ch], h3 = sh3[ch];

  v8f acc[4];
#pragma unroll
  for (int mt = 0; mt < 4; mt++) acc[mt] = vzero8();
#pragma unroll
  for (int kb = 0; kb < 2; kb++) {
    v16bf bfrag = frag_b(&rw_s[ch][0], kb, half);
#pragma unroll
    for (int mt = 0; mt < 4; mt++) {
      v16bf afrag = frag_a(&x_t[mt * 16 + mrow][0], kb, half);
      acc[mt] = wmma_bf16(afrag, bfrag, acc[mt]);
    }
  }
#pragma unroll
  for (int mt = 0; mt < 4; mt++) {
    const int v0 = mt * 16 + half * 8;
    v8bf zv = *(const v8bf*)&z_t[ch][v0];
    float oarr[8];
#pragma unroll
    for (int r = 0; r < 8; r++)
      oarr[r] = fmaxf(acc[mt][r] + rbv + s3 * bf2f(zv[r]) + h3, 0.0f);
    *(float4*)&out_s[ch][v0] = *(const float4*)&oarr[0];
    *(float4*)&out_s[ch][v0 + 4] = *(const float4*)&oarr[4];
  }
  __syncthreads();
  for (int qd = tid; qd < 1024; qd += 128) { // 64*64/4 quads, coalesced
    int r = qd >> 4, c4 = (qd & 15) * 4;
    *(float4*)(out + gbase + (size_t)r * 65536 + c4) =
        *(const float4*)&out_s[r][c4];
  }
}

// ---------------------------------------------------------------------------
extern "C" void kernel_launch(void* const* d_in, const int* in_sizes, int n_in,
                              void* d_out, int out_size, void* d_ws, size_t ws_size,
                              hipStream_t stream) {
  (void)in_sizes; (void)n_in; (void)out_size; (void)ws_size;
  const float* x      = (const float*)d_in[0];
  const float* adj    = (const float*)d_in[1];
  const float* gcn_w  = (const float*)d_in[2];
  const float* bn1_g  = (const float*)d_in[3];
  const float* bn1_b  = (const float*)d_in[4];
  const float* bn2_g  = (const float*)d_in[5];
  const float* bn2_b  = (const float*)d_in[6];
  const float* conv_w = (const float*)d_in[7];
  const float* conv_b = (const float*)d_in[8];
  const float* bn3_g  = (const float*)d_in[9];
  const float* bn3_b  = (const float*)d_in[10];
  const float* res_w  = (const float*)d_in[11];
  const float* res_b  = (const float*)d_in[12];
  float* out = (float*)d_out;
  char* wsb = (char*)d_ws;

  __bf16* hraw = (__bf16*)wsb;                // 33,554,432 bf16 (64 MB)
  __bf16* zbuf = (__bf16*)(wsb + 67108864);   // 33,554,432 bf16 (64 MB)
  float* partial = (float*)(wsb + 134217728); // 65,536 floats
  float* sc1 = partial + 65536;
  float* sh1 = sc1 + 64;
  float* sc2 = sh1 + 64;
  float* sh2 = sc2 + 64;
  float* sc3 = sh2 + 64;
  float* sh3 = sc3 + 64;

  gcn_kernel<<<512, 256, 0, stream>>>(x, adj, gcn_w, hraw, partial);
  stats_finalize<<<1, 64, 0, stream>>>(partial, bn1_g, bn1_b, sc1, sh1);
  stats_partial<<<512, 256, 0, stream>>>(hraw, partial, sc1, sh1);
  stats_finalize<<<1, 64, 0, stream>>>(partial, bn2_g, bn2_b, sc2, sh2);
  conv_kernel<<<512, 256, 0, stream>>>(hraw, conv_w, conv_b, sc1, sh1, sc2, sh2,
                                       zbuf, partial);
  stats_finalize<<<1, 64, 0, stream>>>(partial, bn3_g, bn3_b, sc3, sh3);
  final_kernel<<<8192, 128, 0, stream>>>(x, res_w, res_b, zbuf, sc3, sh3, out);
}
